// MultilHeadAttention_81647328297267
// MI455X (gfx1250) — matrix operand order repaired and checked by reading, not compiled, not run on hardware
//
#include <hip/hip_runtime.h>
#include <hip/hip_bf16.h>
#include <math.h>

#define D_MODEL 512
#define DK      64
#define NH      8
#define NSEQ    4096

typedef __attribute__((ext_vector_type(16))) __bf16 v16bf;
typedef __attribute__((ext_vector_type(8)))  float  v8f;
typedef __attribute__((ext_vector_type(4)))  unsigned int v4u;
typedef __attribute__((ext_vector_type(8)))  int v8i_;
typedef __attribute__((ext_vector_type(4)))  int v4i_;

#define WMMA_BF16(A, B, C) \
  __builtin_amdgcn_wmma_f32_16x16x32_bf16(false, (A), false, (B), (short)0, (C), false, false)

#if __has_builtin(__builtin_amdgcn_tensor_load_to_lds)
#define USE_TDM 1
#else
#define USE_TDM 0
#endif

__device__ __forceinline__ unsigned short f2bf(float f) {
  unsigned int u = __float_as_uint(f);
  unsigned int r = (u + 0x7FFFu + ((u >> 16) & 1u)) >> 16;   // round-to-nearest-even
  return (unsigned short)r;
}

union Frag {
  v16bf v;
  uint4 q[2];
  unsigned short s[16];
};

// A operand: 16(M) x 32(K) bf16 tile, row-major, ld in elements.
__device__ __forceinline__ v16bf load_a(const unsigned short* base, int ld) {
  int lane = threadIdx.x & 31;
  int m = lane & 15, hh = lane >> 4;
  Frag f;
  f.q[0] = *(const uint4*)(base + m * ld + hh * 8);
  f.q[1] = *(const uint4*)(base + m * ld + 16 + hh * 8);
  return f.v;
}

// B operand: 32(K) x 16(N) bf16 tile, row-major, ld in elements. Lane = column n, elements over k (table order).
__device__ __forceinline__ v16bf load_b(const unsigned short* base, int ld) {
  int lane = threadIdx.x & 31;
  int n = lane & 15, hh = lane >> 4;
  Frag f;
#pragma unroll
  for (int s = 0; s < 8; ++s) {
    f.s[s]     = base[(hh * 8 + s) * ld + n];
    f.s[8 + s] = base[(16 + hh * 8 + s) * ld + n];
  }
  return f.v;
}

#if USE_TDM
__device__ __forceinline__ unsigned lds_byte_off(const void* p) {
  return (unsigned)(uintptr_t)(__attribute__((address_space(3))) const void*)p;
}

// 2D tile load via Tensor Data Mover: rows x cols (2-byte elems), row stride in elems.
__device__ __forceinline__ void tdm_load_2d(const void* gsrc, unsigned ldsoff,
                                            unsigned rows, unsigned cols,
                                            unsigned stride_elems) {
  unsigned long long ga = (unsigned long long)(uintptr_t)gsrc;
  v4u g0;
  g0[0] = 1u;                                      // count=1 (valid), user mode
  g0[1] = ldsoff;                                  // lds_addr (bytes)
  g0[2] = (unsigned)(ga & 0xFFFFFFFFu);            // global_addr[31:0]
  g0[3] = (unsigned)((ga >> 32) & 0x01FFFFFFu)     // global_addr[56:32]
          | (2u << 30);                            // type=2 ("image")
  v8i_ g1;
  g1[0] = (int)(1u << 16);                         // wg_mask=0, data_size=1 (2B)
  g1[1] = (int)((cols & 0xFFFFu) << 16);           // tensor_dim0[15:0]
  g1[2] = (int)(((cols >> 16) & 0xFFFFu) | ((rows & 0xFFFFu) << 16)); // dim0 hi | dim1 lo
  g1[3] = (int)(((rows >> 16) & 0xFFFFu) | ((cols & 0xFFFFu) << 16)); // dim1 hi | tile_dim0
  g1[4] = (int)(rows & 0xFFFFu);                   // tile_dim1 | tile_dim2=0
  g1[5] = (int)stride_elems;                       // tensor_dim0_stride[31:0]
  g1[6] = 0;                                       // stride hi | dim1_stride lo
  g1[7] = 0;
  v4i_ z4 = {0, 0, 0, 0};
  v8i_ z8 = {0, 0, 0, 0, 0, 0, 0, 0};
  // 6-arg toolchain form: (g0, g1, g2, g3, g4, cpol)
  __builtin_amdgcn_tensor_load_to_lds(g0, g1, z4, z4, z8, 0);
}
#endif

// ---------------- prep kernels ----------------
__global__ void mha_cvt_bf16(const float* __restrict__ src,
                             unsigned short* __restrict__ dst, int n) {
  int i = blockIdx.x * blockDim.x + threadIdx.x;
  if (i < n) dst[i] = f2bf(src[i]);
}

__global__ void mha_transpose_w(const float* __restrict__ W,
                                unsigned short* __restrict__ Wt) {
  int i = blockIdx.x * blockDim.x + threadIdx.x;   // over 512*512
  int n = i >> 9, k = i & 511;
  Wt[k * 512 + n] = f2bf(W[n * 512 + k]);
}

__global__ void mha_transpose_k(const unsigned short* __restrict__ Kp,
                                unsigned short* __restrict__ Kpt) {
  int i = blockIdx.x * blockDim.x + threadIdx.x;   // over 4096*512
  int nrow = i >> 9;
  int c = i & 511;
  Kpt[c * NSEQ + nrow] = Kp[i];
}

// ------------- projection GEMM: 32x64 per wave, Y(bf16) = X @ Wt + bias -------------
__global__ __launch_bounds__(256) void mha_proj_gemm(
    const unsigned short* __restrict__ X,    // [4096,512] bf16
    const unsigned short* __restrict__ Wt,   // [512,512]  bf16 k-major
    const float* __restrict__ bias,
    unsigned short* __restrict__ Y)
{
  int wave = (int)((blockIdx.x * blockDim.x + threadIdx.x) >> 5);
  int lane = threadIdx.x & 31;
  int tm = wave >> 3;          // 0..127
  int tn = wave & 7;           // 0..7
  int m0 = tm * 32, n0 = tn * 64;
  v8f c[2][4];
#pragma unroll
  for (int i = 0; i < 2; ++i)
#pragma unroll
    for (int j = 0; j < 4; ++j) c[i][j] = (v8f){};

  for (int k = 0; k < D_MODEL; k += 32) {
    v16bf a0 = load_a(X + (m0 +  0) * D_MODEL + k, D_MODEL);
    v16bf a1 = load_a(X + (m0 + 16) * D_MODEL + k, D_MODEL);
    v16bf b0 = load_b(Wt + k * D_MODEL + n0 +  0, D_MODEL);
    v16bf b1 = load_b(Wt + k * D_MODEL + n0 + 16, D_MODEL);
    v16bf b2 = load_b(Wt + k * D_MODEL + n0 + 32, D_MODEL);
    v16bf b3 = load_b(Wt + k * D_MODEL + n0 + 48, D_MODEL);
    c[0][0] = WMMA_BF16(a0, b0, c[0][0]);
    c[0][1] = WMMA_BF16(a0, b1, c[0][1]);
    c[0][2] = WMMA_BF16(a0, b2, c[0][2]);
    c[0][3] = WMMA_BF16(a0, b3, c[0][3]);
    c[1][0] = WMMA_BF16(a1, b0, c[1][0]);
    c[1][1] = WMMA_BF16(a1, b1, c[1][1]);
    c[1][2] = WMMA_BF16(a1, b2, c[1][2]);
    c[1][3] = WMMA_BF16(a1, b3, c[1][3]);
  }
  int hh = lane >> 4, nlow = lane & 15;
#pragma unroll
  for (int j = 0; j < 4; ++j) {
    int col = n0 + 16 * j + nlow;
    float bv = bias[col];
#pragma unroll
    for (int i = 0; i < 2; ++i) {
#pragma unroll
      for (int r = 0; r < 8; ++r) {
        int row = m0 + 16 * i + r + 8 * hh;
        Y[row * D_MODEL + col] = f2bf(c[i][j][r] + bv);
      }
    }
  }
}

// ---------------- flash attention (one head per block, TDM-staged K/V) ----------------
__global__ __launch_bounds__(256) void mha_flash_attn(
    const unsigned short* __restrict__ Qp,   // [4096,512] bf16
    const unsigned short* __restrict__ Kpt,  // [512,4096] bf16 (h*64+d major)
    const unsigned short* __restrict__ Vp,   // [4096,512] bf16
    unsigned short* __restrict__ CTX)        // [4096,512] bf16
{
  __shared__ __align__(16) unsigned short Kt[64 * 32];      // [d][key]   4 KB
  __shared__ __align__(16) unsigned short Vt[32 * 64];      // [key][dv]  4 KB
  __shared__ __align__(16) unsigned short plds[8][16 * 32]; // per-wave P 8 KB
  int wib  = threadIdx.x >> 5;
  int lane = threadIdx.x & 31;
  int h  = blockIdx.x >> 5;                 // 8 heads
  int qt = (blockIdx.x & 31) * 8 + wib;     // 256 q-tiles
  int m0 = qt * 16;
  int lh   = lane >> 4;
  int nlow = lane & 15;

  v16bf aq0 = load_a(Qp + m0 * D_MODEL + h * DK + 0,  D_MODEL);
  v16bf aq1 = load_a(Qp + m0 * D_MODEL + h * DK + 32, D_MODEL);

  float rowmax[8], rowsum[8];
  v8f o0 = {}, o1 = {}, o2 = {}, o3 = {};
#pragma unroll
  for (int r = 0; r < 8; ++r) { rowmax[r] = -__builtin_inff(); rowsum[r] = 0.f; }

  unsigned short* lp = plds[wib];
  const unsigned short* kbase = Kpt + h * DK * NSEQ;

  for (int n0 = 0; n0 < NSEQ; n0 += 32) {
    __syncthreads();   // all waves done reading previous K/V tiles
#if USE_TDM
    if (wib == 0) {
      tdm_load_2d(kbase + n0, lds_byte_off(Kt), 64, 32, NSEQ);
      tdm_load_2d(Vp + n0 * D_MODEL + h * DK, lds_byte_off(Vt), 32, 64, D_MODEL);
      __builtin_amdgcn_s_wait_tensorcnt((short)0);
    }
#else
    {
      int t = threadIdx.x;
      int kr = t >> 2, kq = t & 3;          // 64 rows x 4 chunks of 8 elems
      *(uint4*)(Kt + kr * 32 + kq * 8) =
          *(const uint4*)(kbase + kr * NSEQ + n0 + kq * 8);
      int vr = t >> 3, vq = t & 7;          // 32 rows x 8 chunks of 8 elems
      *(uint4*)(Vt + vr * 64 + vq * 8) =
          *(const uint4*)(Vp + (n0 + vr) * D_MODEL + h * DK + vq * 8);
    }
#endif
    __syncthreads();   // tiles ready

    // scores S = (Q·K^T)/8, two 16x16 D tiles, B-frags from LDS
    v8f s0 = {}, s1 = {};
    s0 = WMMA_BF16(aq0, load_b(Kt +  0 * 32 +  0, 32), s0);
    s0 = WMMA_BF16(aq1, load_b(Kt + 32 * 32 +  0, 32), s0);
    s1 = WMMA_BF16(aq0, load_b(Kt +  0 * 32 + 16, 32), s1);
    s1 = WMMA_BF16(aq1, load_b(Kt + 32 * 32 + 16, 32), s1);

#pragma unroll
    for (int r = 0; r < 8; ++r) {
      float a = s0[r] * 0.125f;
      float b = s1[r] * 0.125f;
      float mx = fmaxf(a, b);
      for (int mk = 1; mk < 16; mk <<= 1) mx = fmaxf(mx, __shfl_xor(mx, mk, 32));
      float nm   = fmaxf(rowmax[r], mx);
      float corr = __expf(rowmax[r] - nm);
      float p0 = __expf(a - nm);
      float p1 = __expf(b - nm);
      float sum = p0 + p1;
      for (int mk = 1; mk < 16; mk <<= 1) sum += __shfl_xor(sum, mk, 32);
      rowsum[r] = rowsum[r] * corr + sum;
      rowmax[r] = nm;
      o0[r] *= corr; o1[r] *= corr; o2[r] *= corr; o3[r] *= corr;
      s0[r] = p0; s1[r] = p1;   // now P
    }

    // P (D layout) -> LDS -> A layout
#pragma unroll
    for (int r = 0; r < 8; ++r) {
      int row = r + 8 * lh;
      lp[row * 32 + nlow]      = f2bf(s0[r]);
      lp[row * 32 + 16 + nlow] = f2bf(s1[r]);
    }
    asm volatile("s_wait_dscnt 0" ::: "memory");
    Frag fp;
    fp.q[0] = *(const uint4*)(lp + nlow * 32 + lh * 8);
    fp.q[1] = *(const uint4*)(lp + nlow * 32 + 16 + lh * 8);

    // O += P @ V (B-frags from LDS V tile)
    o0 = WMMA_BF16(fp.v, load_b(Vt +  0, 64), o0);
    o1 = WMMA_BF16(fp.v, load_b(Vt + 16, 64), o1);
    o2 = WMMA_BF16(fp.v, load_b(Vt + 32, 64), o2);
    o3 = WMMA_BF16(fp.v, load_b(Vt + 48, 64), o3);
  }

#pragma unroll
  for (int r = 0; r < 8; ++r) {
    float inv = 1.0f / rowsum[r];
    int row = m0 + r + 8 * lh;
    unsigned short* cp = CTX + row * D_MODEL + h * DK;
    cp[nlow]      = f2bf(o0[r] * inv);
    cp[16 + nlow] = f2bf(o1[r] * inv);
    cp[32 + nlow] = f2bf(o2[r] * inv);
    cp[48 + nlow] = f2bf(o3[r] * inv);
  }
}

// ------------- output GEMM: 32x64 per wave, tmp = CTX @ WOt + bO + Q -------------
__global__ __launch_bounds__(256) void mha_out_gemm(
    const unsigned short* __restrict__ CTX,
    const unsigned short* __restrict__ WOt,
    const float* __restrict__ bias,
    const float* __restrict__ Qres,
    float* __restrict__ Y)
{
  int wave = (int)((blockIdx.x * blockDim.x + threadIdx.x) >> 5);
  int lane = threadIdx.x & 31;
  int tm = wave >> 3;
  int tn = wave & 7;
  int m0 = tm * 32, n0 = tn * 64;
  v8f c[2][4];
#pragma unroll
  for (int i = 0; i < 2; ++i)
#pragma unroll
    for (int j = 0; j < 4; ++j) c[i][j] = (v8f){};

  for (int k = 0; k < D_MODEL; k += 32) {
    v16bf a0 = load_a(CTX + (m0 +  0) * D_MODEL + k, D_MODEL);
    v16bf a1 = load_a(CTX + (m0 + 16) * D_MODEL + k, D_MODEL);
    v16bf b0 = load_b(WOt + k * D_MODEL + n0 +  0, D_MODEL);
    v16bf b1 = load_b(WOt + k * D_MODEL + n0 + 16, D_MODEL);
    v16bf b2 = load_b(WOt + k * D_MODEL + n0 + 32, D_MODEL);
    v16bf b3 = load_b(WOt + k * D_MODEL + n0 + 48, D_MODEL);
    c[0][0] = WMMA_BF16(a0, b0, c[0][0]);
    c[0][1] = WMMA_BF16(a0, b1, c[0][1]);
    c[0][2] = WMMA_BF16(a0, b2, c[0][2]);
    c[0][3] = WMMA_BF16(a0, b3, c[0][3]);
    c[1][0] = WMMA_BF16(a1, b0, c[1][0]);
    c[1][1] = WMMA_BF16(a1, b1, c[1][1]);
    c[1][2] = WMMA_BF16(a1, b2, c[1][2]);
    c[1][3] = WMMA_BF16(a1, b3, c[1][3]);
  }
  int hh = lane >> 4, nlow = lane & 15;
#pragma unroll
  for (int j = 0; j < 4; ++j) {
    int col = n0 + 16 * j + nlow;
    float bv = bias[col];
#pragma unroll
    for (int i = 0; i < 2; ++i) {
#pragma unroll
      for (int r = 0; r < 8; ++r) {
        int row = m0 + 16 * i + r + 8 * hh;
        Y[row * D_MODEL + col] = c[i][j][r] + bv + Qres[row * D_MODEL + col];
      }
    }
  }
}

// ---------------- LayerNorm: one wave per row of 512 ----------------
__global__ __launch_bounds__(256) void mha_layernorm(
    const float* __restrict__ X, const float* __restrict__ gamma,
    const float* __restrict__ beta, float* __restrict__ Y)
{
  int wave = (int)((blockIdx.x * blockDim.x + threadIdx.x) >> 5);
  int lane = threadIdx.x & 31;
  const float* row = X + (size_t)wave * D_MODEL;
  float v[16];
  float s = 0.f;
#pragma unroll
  for (int i = 0; i < 16; ++i) { v[i] = row[lane * 16 + i]; s += v[i]; }
  for (int mk = 1; mk < 32; mk <<= 1) s += __shfl_xor(s, mk, 32);
  float mu = s * (1.0f / D_MODEL);
  float var = 0.f;
#pragma unroll
  for (int i = 0; i < 16; ++i) { float d = v[i] - mu; var += d * d; }
  for (int mk = 1; mk < 32; mk <<= 1) var += __shfl_xor(var, mk, 32);
  var *= (1.0f / D_MODEL);
  float rstd = rsqrtf(var + 1e-5f);
#pragma unroll
  for (int i = 0; i < 16; ++i) {
    int cidx = lane * 16 + i;
    Y[(size_t)wave * D_MODEL + cidx] = (v[i] - mu) * rstd * gamma[cidx] + beta[cidx];
  }
}

extern "C" void kernel_launch(void* const* d_in, const int* in_sizes, int n_in,
                              void* d_out, int out_size, void* d_ws, size_t ws_size,
                              hipStream_t stream) {
  (void)in_sizes; (void)n_in; (void)out_size; (void)ws_size;
  const float* Q  = (const float*)d_in[0];
  const float* K  = (const float*)d_in[1];
  const float* V  = (const float*)d_in[2];
  const float* WQ = (const float*)d_in[3];
  const float* bQ = (const float*)d_in[4];
  const float* WK = (const float*)d_in[5];
  const float* bK = (const float*)d_in[6];
  const float* WV = (const float*)d_in[7];
  const float* bV = (const float*)d_in[8];
  const float* WO = (const float*)d_in[9];
  const float* bO = (const float*)d_in[10];
  const float* gamma = (const float*)d_in[11];
  const float* beta  = (const float*)d_in[12];
  float* out = (float*)d_out;

  size_t off = 0;
  auto carve = [&](size_t bytes) -> void* {
    void* p = (char*)d_ws + off;
    off += (bytes + 255) & ~(size_t)255;
    return p;
  };
  const size_t ACT = (size_t)NSEQ * D_MODEL;
  const size_t WEL = (size_t)D_MODEL * D_MODEL;
  unsigned short* Qb  = (unsigned short*)carve(ACT * 2);
  unsigned short* Kb  = (unsigned short*)carve(ACT * 2);
  unsigned short* Vb  = (unsigned short*)carve(ACT * 2);
  unsigned short* WQt = (unsigned short*)carve(WEL * 2);
  unsigned short* WKt = (unsigned short*)carve(WEL * 2);
  unsigned short* WVt = (unsigned short*)carve(WEL * 2);
  unsigned short* WOt = (unsigned short*)carve(WEL * 2);
  unsigned short* Qp  = (unsigned short*)carve(ACT * 2);
  unsigned short* Kp  = (unsigned short*)carve(ACT * 2);
  unsigned short* Vp  = (unsigned short*)carve(ACT * 2);
  unsigned short* Kpt = (unsigned short*)carve(ACT * 2);
  unsigned short* CTX = (unsigned short*)carve(ACT * 2);
  float*          TMP = (float*)carve(ACT * 4);

  dim3 blk(256);
  mha_cvt_bf16<<<dim3(ACT / 256), blk, 0, stream>>>(Q, Qb, (int)ACT);
  mha_cvt_bf16<<<dim3(ACT / 256), blk, 0, stream>>>(K, Kb, (int)ACT);
  mha_cvt_bf16<<<dim3(ACT / 256), blk, 0, stream>>>(V, Vb, (int)ACT);
  mha_transpose_w<<<dim3(WEL / 256), blk, 0, stream>>>(WQ, WQt);
  mha_transpose_w<<<dim3(WEL / 256), blk, 0, stream>>>(WK, WKt);
  mha_transpose_w<<<dim3(WEL / 256), blk, 0, stream>>>(WV, WVt);
  mha_transpose_w<<<dim3(WEL / 256), blk, 0, stream>>>(WO, WOt);
  // 1024 waves each: 128 tiles(M=32) x 8 tiles(N=64), 8 waves/block
  mha_proj_gemm<<<dim3(128), blk, 0, stream>>>(Qb, WQt, bQ, Qp);
  mha_proj_gemm<<<dim3(128), blk, 0, stream>>>(Kb, WKt, bK, Kp);
  mha_proj_gemm<<<dim3(128), blk, 0, stream>>>(Vb, WVt, bV, Vp);
  mha_transpose_k<<<dim3(ACT / 256), blk, 0, stream>>>(Kp, Kpt);
  // 8 heads x 32 q-groups; block = one head, 8 q-tiles
  mha_flash_attn<<<dim3(256), blk, 0, stream>>>(Qp, Kpt, Vp, CTX);
  mha_out_gemm<<<dim3(128), blk, 0, stream>>>(CTX, WOt, bO, Q, TMP);
  mha_layernorm<<<dim3(512), blk, 0, stream>>>(TMP, gamma, beta, out);
}
